// RegionProposalNetwork_64527588655671
// MI455X (gfx1250) — matrix operand order repaired and checked by reading, not compiled, not run on hardware
//
#include <hip/hip_runtime.h>
#include <math.h>

typedef unsigned short u16;
typedef __attribute__((ext_vector_type(8)))  u16    u16x8;
typedef __attribute__((ext_vector_type(16))) u16    u16x16;
typedef __attribute__((ext_vector_type(16))) __bf16 v16bf;
typedef __attribute__((ext_vector_type(8)))  float  v8f;

union FragAB {
  u16x16 u;
  u16x8  h[2];
  v16bf  bf;
};

__device__ __forceinline__ u16 f2bf(float f) {
  unsigned u = __float_as_uint(f);
  u += 0x7FFFu + ((u >> 16) & 1u);   // round-to-nearest-even
  return (u16)(u >> 16);
}
__device__ __forceinline__ float bf2f(u16 h) {
  return __uint_as_float(((unsigned)h) << 16);
}

// ---------------------------------------------------------------------------
// Prep: conv_w [256,256,3,3] fp32 -> Wb bf16 [Cout=256, K=2304], K = (kh*3+kw)*256 + cin
// ---------------------------------------------------------------------------
__global__ void prep_weights(const float* __restrict__ w, u16* __restrict__ Wb) {
  int tid = blockIdx.x * blockDim.x + threadIdx.x;
  if (tid >= 256 * 2304) return;
  int m   = tid / 2304;
  int k   = tid - m * 2304;
  int p   = k >> 8;          // tap index 0..8
  int cin = k & 255;
  int kh = p / 3, kw = p - kh * 3;
  Wb[tid] = f2bf(w[((m * 256 + cin) * 3 + kh) * 3 + kw]);
}

// ---------------------------------------------------------------------------
// Prep: feat NCHW fp32 -> zero-padded NHWC bf16 [bs, H+2, W+2, 256]
// ---------------------------------------------------------------------------
__global__ void prep_feat(const float* __restrict__ f, u16* __restrict__ fb,
                          int H, int W) {
  int tid = blockIdx.x * blockDim.x + threadIdx.x;
  int total = 4 * (H + 2) * (W + 2) * 256;
  if (tid >= total) return;
  int c  = tid & 255;
  int r  = tid >> 8;
  int xp = r % (W + 2); r /= (W + 2);
  int yp = r % (H + 2);
  int bs = r / (H + 2);
  float v = 0.f;
  if (xp >= 1 && xp <= W && yp >= 1 && yp <= H)
    v = f[(((size_t)bs * 256 + c) * H + (yp - 1)) * W + (xp - 1)];
  fb[tid] = f2bf(v);
}

// ---------------------------------------------------------------------------
// 3x3 conv + bias + ReLU as implicit GEMM with v_wmma_f32_16x16x32_bf16.
// 2x2 register blocking: one wave computes 32(cout) x 32(pixel) with 4
// independent accumulators; each K-chunk reuses both A fragments against both
// B fragments -> 8 b128 loads per 4 wmma and no wmma->wmma RAW hazards.
// A (weights 16x32): lane m=lane&15, hi=lane>>4: slots = K[hi*16..+16) contiguous
// B (im2col 32x16): lane col n=lane&15, hi=lane>>4: slots = K[hi*16..+16) contiguous cin
// D: VGPR g -> cout = mbase + 8*hi + g, col n -> 8 contiguous bf16 NHWC store
// ---------------------------------------------------------------------------
__global__ void __launch_bounds__(32) conv3x3_relu_wmma(
    const u16* __restrict__ Wb, const u16* __restrict__ fb,
    const float* __restrict__ bias, u16* __restrict__ fx, int H, int W) {
  const int lane  = threadIdx.x;
  const int n     = lane & 15;
  const int hi    = lane >> 4;
  const int mbase = blockIdx.y << 5;   // 32 couts per block
  const int nbase = blockIdx.x << 5;   // 32 pixels per block
  const int bs    = blockIdx.z;
  const int W2    = W + 2;

  const int pos0 = nbase + n;
  const int pos1 = nbase + 16 + n;
  const int h0 = pos0 / W, w0 = pos0 - h0 * W;
  const int h1 = pos1 / W, w1 = pos1 - h1 * W;

  const u16* fbb   = fb + (size_t)bs * (H + 2) * W2 * 256;
  const u16* arow0 = Wb + (size_t)(mbase + n) * 2304;
  const u16* arow1 = arow0 + (size_t)16 * 2304;

  v8f acc00 = {}, acc01 = {}, acc10 = {}, acc11 = {};
  for (int kk = 0; kk < 72; ++kk) {
    const int p    = kk >> 3;
    const int cinb = (kk & 7) << 5;
    const int kh = p / 3, kw = p - kh * 3;
    const int kbase = (p << 8) + cinb;

    FragAB A0, A1, B0, B1;
    A0.h[0] = *(const u16x8*)(arow0 + kbase + hi * 16);
    A0.h[1] = *(const u16x8*)(arow0 + kbase + 8 + hi * 16);
    A1.h[0] = *(const u16x8*)(arow1 + kbase + hi * 16);
    A1.h[1] = *(const u16x8*)(arow1 + kbase + 8 + hi * 16);
    const u16* bp0 = fbb + (((size_t)(h0 + kh) * W2 + (w0 + kw)) << 8) + cinb + (hi << 4);
    const u16* bp1 = fbb + (((size_t)(h1 + kh) * W2 + (w1 + kw)) << 8) + cinb + (hi << 4);
    B0.h[0] = *(const u16x8*)(bp0);
    B0.h[1] = *(const u16x8*)(bp0 + 8);
    B1.h[0] = *(const u16x8*)(bp1);
    B1.h[1] = *(const u16x8*)(bp1 + 8);

    // pull the next K-chunk's weight rows toward the WGP (global_prefetch_b8)
    if (kk + 1 < 72) {
      __builtin_prefetch(arow0 + kbase + 32, 0, 3);
      __builtin_prefetch(arow1 + kbase + 32, 0, 3);
    }

    acc00 = __builtin_amdgcn_wmma_f32_16x16x32_bf16(false, A0.bf, false, B0.bf,
                                                    (short)0, acc00, false, false);
    acc01 = __builtin_amdgcn_wmma_f32_16x16x32_bf16(false, A0.bf, false, B1.bf,
                                                    (short)0, acc01, false, false);
    acc10 = __builtin_amdgcn_wmma_f32_16x16x32_bf16(false, A1.bf, false, B0.bf,
                                                    (short)0, acc10, false, false);
    acc11 = __builtin_amdgcn_wmma_f32_16x16x32_bf16(false, A1.bf, false, B1.bf,
                                                    (short)0, acc11, false, false);
  }

  const int cb0 = mbase + (hi << 3);
  const int cb1 = cb0 + 16;
  float b0v[8], b1v[8];
#pragma unroll
  for (int g = 0; g < 8; ++g) { b0v[g] = bias[cb0 + g]; b1v[g] = bias[cb1 + g]; }

  u16x8 o00, o01, o10, o11;
#pragma unroll
  for (int g = 0; g < 8; ++g) {
    float v00 = acc00[g] + b0v[g];
    float v01 = acc01[g] + b0v[g];
    float v10 = acc10[g] + b1v[g];
    float v11 = acc11[g] + b1v[g];
    o00[g] = f2bf(v00 > 0.f ? v00 : 0.f);
    o01[g] = f2bf(v01 > 0.f ? v01 : 0.f);
    o10[g] = f2bf(v10 > 0.f ? v10 : 0.f);
    o11[g] = f2bf(v11 > 0.f ? v11 : 0.f);
  }
  u16* fx0 = fx + (((size_t)bs * H * W + pos0) << 8);
  u16* fx1 = fx + (((size_t)bs * H * W + pos1) << 8);
  *(u16x8*)(fx0 + cb0) = o00;
  *(u16x8*)(fx1 + cb0) = o01;
  *(u16x8*)(fx0 + cb1) = o10;
  *(u16x8*)(fx1 + cb1) = o11;
}

// ---------------------------------------------------------------------------
// 1x1 heads: per (bs,pos,anchor): cls dot + sigmoid, 4 reg dots + box decode/clamp
// ---------------------------------------------------------------------------
__global__ void __launch_bounds__(256) heads_kernel(
    const u16* __restrict__ fx,
    const float* __restrict__ cls_w, const float* __restrict__ cls_b,
    const float* __restrict__ reg_w, const float* __restrict__ reg_b,
    float* __restrict__ boxes, float* __restrict__ scores,
    int H, int W, float stride, float scale) {
  int tid = blockIdx.x * blockDim.x + threadIdx.x;
  int HW = H * W;
  int total = 4 * HW * 3;
  if (tid >= total) return;
  int a   = tid % 3;
  int pos = (tid / 3) % HW;
  int bs  = tid / (3 * HW);

  const u16* xp = fx + (((size_t)bs * HW + pos) << 8);
  float s0 = cls_b[a];
  float d0 = reg_b[a * 4 + 0], d1 = reg_b[a * 4 + 1];
  float d2 = reg_b[a * 4 + 2], d3 = reg_b[a * 4 + 3];
  const float* cw = cls_w + a * 256;
  const float* rw = reg_w + a * 4 * 256;
  for (int c = 0; c < 256; ++c) {
    float x = bf2f(xp[c]);
    s0 += x * cw[c];
    d0 += x * rw[c];
    d1 += x * rw[256 + c];
    d2 += x * rw[512 + c];
    d3 += x * rw[768 + c];
  }
  float score = 1.f / (1.f + __expf(-s0));

  int h = pos / W, w = pos - h * W;
  float ratio = (a == 0) ? 0.5f : (a == 1 ? 1.f : 2.f);
  float ws = sqrtf(scale / ratio), hs = sqrtf(scale * ratio);
  float cx = ((float)w + 0.5f) * stride, cy = ((float)h + 0.5f) * stride;
  float ax1 = fminf(fmaxf(cx - 0.5f * ws, 0.f), 256.f);
  float ay1 = fminf(fmaxf(cy - 0.5f * hs, 0.f), 256.f);
  float ax2 = fminf(fmaxf(cx + 0.5f * ws, 0.f), 256.f);
  float ay2 = fminf(fmaxf(cy + 0.5f * hs, 0.f), 256.f);
  float wa = ax2 - ax1, ha = ay2 - ay1;
  float cxa = ax1 + 0.5f * wa, cya = ay1 + 0.5f * ha;
  float bcx = d0 * wa + cxa, bcy = d1 * ha + cya;
  float bw = __expf(d2) * wa, bh = __expf(d3) * ha;
  float x1 = fminf(fmaxf(bcx - 0.5f * bw, 0.f), 256.f);
  float y1 = fminf(fmaxf(bcy - 0.5f * bh, 0.f), 256.f);
  float x2 = fminf(fmaxf(bcx + 0.5f * bw, 0.f), 256.f);
  float y2 = fminf(fmaxf(bcy + 0.5f * bh, 0.f), 256.f);

  size_t oi = (size_t)bs * HW * 3 + (size_t)pos * 3 + a;
  boxes[oi * 4 + 0] = x1; boxes[oi * 4 + 1] = y1;
  boxes[oi * 4 + 2] = x2; boxes[oi * 4 + 3] = y2;
  scores[oi] = score;
}

// ---------------------------------------------------------------------------
// Exact greedy NMS per (bs): LDS bitonic sort desc by score, then sequential
// keep loop with block-parallel suppression; early exit at topk kept; -1 pads.
// ---------------------------------------------------------------------------
__global__ void __launch_bounds__(1024) nms_kernel(
    const float* __restrict__ boxes, const float* __restrict__ scores,
    int N, int Npad, float* __restrict__ sbox,
    float* __restrict__ out_b, float* __restrict__ out_s,
    int topk, int out_stride) {
  __shared__ float s_sc[4096];
  __shared__ int   s_ix[4096];
  __shared__ unsigned char s_supp[4096];
  __shared__ int s_cnt;
  const int tid = threadIdx.x, bd = blockDim.x;
  const int bs = blockIdx.x;
  const float* bx = boxes  + (size_t)bs * N * 4;
  const float* sc = scores + (size_t)bs * N;
  float* sb = sbox  + (size_t)bs * N * 4;
  float* ob = out_b + (size_t)bs * out_stride * 4;
  float* os = out_s + (size_t)bs * out_stride;

  for (int k = tid; k < Npad; k += bd) {
    s_sc[k] = (k < N) ? sc[k] : -3.0e38f;
    s_ix[k] = k;
    s_supp[k] = 0;
  }
  for (int k = tid; k < topk; k += bd) {
    os[k] = -1.f;
    ob[k * 4 + 0] = -1.f; ob[k * 4 + 1] = -1.f;
    ob[k * 4 + 2] = -1.f; ob[k * 4 + 3] = -1.f;
  }
  if (tid == 0) s_cnt = 0;
  __syncthreads();

  // bitonic sort, final order descending
  for (int k2 = 2; k2 <= Npad; k2 <<= 1) {
    for (int j = k2 >> 1; j > 0; j >>= 1) {
      for (int t = tid; t < Npad; t += bd) {
        int x = t ^ j;
        if (x > t) {
          bool desc = ((t & k2) == 0);
          float a = s_sc[t], b = s_sc[x];
          if (desc ? (a < b) : (a > b)) {
            s_sc[t] = b; s_sc[x] = a;
            int ia = s_ix[t]; s_ix[t] = s_ix[x]; s_ix[x] = ia;
          }
        }
      }
      __syncthreads();
    }
  }

  // stage sorted boxes to global scratch (L2-resident)
  for (int k = tid; k < N; k += bd) {
    int src = s_ix[k];
    sb[k * 4 + 0] = bx[src * 4 + 0];
    sb[k * 4 + 1] = bx[src * 4 + 1];
    sb[k * 4 + 2] = bx[src * 4 + 2];
    sb[k * 4 + 3] = bx[src * 4 + 3];
  }
  __threadfence_block();
  __syncthreads();

  for (int i = 0; i < N; ++i) {
    int cnt = s_cnt;               // uniform: updated only between barriers
    if (cnt >= topk) break;
    bool keep = (s_supp[i] == 0);  // uniform
    if (keep) {
      float ix1 = sb[i * 4 + 0], iy1 = sb[i * 4 + 1];
      float ix2 = sb[i * 4 + 2], iy2 = sb[i * 4 + 3];
      if (tid == 0) {
        os[cnt] = s_sc[i];
        ob[cnt * 4 + 0] = ix1; ob[cnt * 4 + 1] = iy1;
        ob[cnt * 4 + 2] = ix2; ob[cnt * 4 + 3] = iy2;
      }
      float iarea = (ix2 - ix1) * (iy2 - iy1);
      for (int jj = i + 1 + tid; jj < N; jj += bd) {
        if (s_supp[jj]) continue;
        float jx1 = sb[jj * 4 + 0], jy1 = sb[jj * 4 + 1];
        float jx2 = sb[jj * 4 + 2], jy2 = sb[jj * 4 + 3];
        float iw = fminf(ix2, jx2) - fmaxf(ix1, jx1);
        float ih = fminf(iy2, jy2) - fmaxf(iy1, jy1);
        iw = iw > 0.f ? iw : 0.f;
        ih = ih > 0.f ? ih : 0.f;
        float inter = iw * ih;
        float ja = (jx2 - jx1) * (jy2 - jy1);
        float iou = inter / (iarea + ja - inter);
        if (iou > 0.7f) s_supp[jj] = 1;
      }
    }
    __syncthreads();
    if (keep && tid == 0) s_cnt = cnt + 1;
    __syncthreads();
  }
}

// ---------------------------------------------------------------------------
// Final: per image, sort concatenated 2000 (scores,-1 pads) desc, gather boxes
// ---------------------------------------------------------------------------
__global__ void __launch_bounds__(512) merge_sort_kernel(
    const float* __restrict__ nmsb, const float* __restrict__ nmss,
    float* __restrict__ out) {
  __shared__ float s_sc[2048];
  __shared__ int   s_ix[2048];
  const int tid = threadIdx.x, bd = blockDim.x;
  const int bs = blockIdx.x;
  const float* sc = nmss + (size_t)bs * 2000;
  const float* bx = nmsb + (size_t)bs * 2000 * 4;
  for (int k = tid; k < 2048; k += bd) {
    s_sc[k] = (k < 2000) ? sc[k] : -3.0e38f;
    s_ix[k] = k;
  }
  __syncthreads();
  for (int k2 = 2; k2 <= 2048; k2 <<= 1) {
    for (int j = k2 >> 1; j > 0; j >>= 1) {
      for (int t = tid; t < 2048; t += bd) {
        int x = t ^ j;
        if (x > t) {
          bool desc = ((t & k2) == 0);
          float a = s_sc[t], b = s_sc[x];
          if (desc ? (a < b) : (a > b)) {
            s_sc[t] = b; s_sc[x] = a;
            int ia = s_ix[t]; s_ix[t] = s_ix[x]; s_ix[x] = ia;
          }
        }
      }
      __syncthreads();
    }
  }
  for (int k = tid; k < 2000; k += bd) {
    int s = s_ix[k];
    float* o = out + ((size_t)bs * 2000 + k) * 4;
    o[0] = bx[s * 4 + 0]; o[1] = bx[s * 4 + 1];
    o[2] = bx[s * 4 + 2]; o[3] = bx[s * 4 + 3];
  }
}

// ---------------------------------------------------------------------------
extern "C" void kernel_launch(void* const* d_in, const int* in_sizes, int n_in,
                              void* d_out, int out_size, void* d_ws, size_t ws_size,
                              hipStream_t stream) {
  (void)in_sizes; (void)n_in; (void)out_size; (void)ws_size;
  const float* feat0  = (const float*)d_in[0];
  const float* feat1  = (const float*)d_in[1];
  const float* conv_w = (const float*)d_in[2];
  const float* conv_b = (const float*)d_in[3];
  const float* cls_w  = (const float*)d_in[4];
  const float* cls_b  = (const float*)d_in[5];
  const float* reg_w  = (const float*)d_in[6];
  const float* reg_b  = (const float*)d_in[7];

  char* ws = (char*)d_ws;
  size_t off = 0;
  auto alloc = [&](size_t bytes) -> void* {
    void* p = ws + off;
    off = (off + bytes + 255) & ~(size_t)255;
    return p;
  };
  u16*   Wb   = (u16*)alloc((size_t)256 * 2304 * 2);
  u16*   fb0  = (u16*)alloc((size_t)4 * 34 * 34 * 256 * 2);
  u16*   fb1  = (u16*)alloc((size_t)4 * 18 * 18 * 256 * 2);
  u16*   fx0  = (u16*)alloc((size_t)4 * 1024 * 256 * 2);
  u16*   fx1  = (u16*)alloc((size_t)4 * 256 * 256 * 2);
  float* bx0  = (float*)alloc((size_t)4 * 3072 * 4 * 4);
  float* sc0  = (float*)alloc((size_t)4 * 3072 * 4);
  float* bx1  = (float*)alloc((size_t)4 * 768 * 4 * 4);
  float* sc1  = (float*)alloc((size_t)4 * 768 * 4);
  float* sb0  = (float*)alloc((size_t)4 * 3072 * 4 * 4);
  float* sb1  = (float*)alloc((size_t)4 * 768 * 4 * 4);
  float* nmsb = (float*)alloc((size_t)4 * 2000 * 4 * 4);
  float* nmss = (float*)alloc((size_t)4 * 2000 * 4);

  prep_weights<<<(256 * 2304 + 255) / 256, 256, 0, stream>>>(conv_w, Wb);
  prep_feat<<<(4 * 34 * 34 * 256 + 255) / 256, 256, 0, stream>>>(feat0, fb0, 32, 32);
  prep_feat<<<(4 * 18 * 18 * 256 + 255) / 256, 256, 0, stream>>>(feat1, fb1, 16, 16);

  conv3x3_relu_wmma<<<dim3(32, 8, 4), 32, 0, stream>>>(Wb, fb0, conv_b, fx0, 32, 32);
  conv3x3_relu_wmma<<<dim3(8, 8, 4), 32, 0, stream>>>(Wb, fb1, conv_b, fx1, 16, 16);

  heads_kernel<<<(4 * 1024 * 3 + 255) / 256, 256, 0, stream>>>(
      fx0, cls_w, cls_b, reg_w, reg_b, bx0, sc0, 32, 32, 8.f, 4096.f);
  heads_kernel<<<(4 * 256 * 3 + 255) / 256, 256, 0, stream>>>(
      fx1, cls_w, cls_b, reg_w, reg_b, bx1, sc1, 16, 16, 16.f, 16384.f);

  nms_kernel<<<4, 1024, 0, stream>>>(bx0, sc0, 3072, 4096, sb0,
                                     nmsb, nmss, 1000, 2000);
  nms_kernel<<<4, 1024, 0, stream>>>(bx1, sc1, 768, 1024, sb1,
                                     nmsb + 1000 * 4, nmss + 1000, 1000, 2000);

  merge_sort_kernel<<<4, 512, 0, stream>>>(nmsb, nmss, (float*)d_out);
}
